// Model_49366354100278
// MI455X (gfx1250) — matrix operand order repaired and checked by reading, not compiled, not run on hardware
//
#include <hip/hip_runtime.h>
#include <hip/hip_bf16.h>
#include <stdint.h>

typedef __attribute__((ext_vector_type(16))) _Float16 v16h;
typedef __attribute__((ext_vector_type(8)))  _Float16 v8h;
typedef __attribute__((ext_vector_type(8)))  float    v8f;

// ---------------- model dims ----------------
constexpr long long cB = 16, cT = 256, cN = 128, cC = 64, cH = 8, cL = 4;
constexpr long long cBTN = cB * cT * cN;          // 524288 rows
constexpr long long cXE  = cBTN * cC;             // 33554432 elements of x

// ---------------- workspace layout (bytes) ----------------
constexpr long long OFF_X   = 0;                      // x   fp32 [B,T,N,C]
constexpr long long OFF_X1  = OFF_X   + cXE * 4;      // x1  fp32
constexpr long long OFF_XH  = OFF_X1  + cXE * 4;      // layernorm out, f16
constexpr long long OFF_V   = OFF_XH  + cXE * 2;      // v_std f16 [B,T,N,C]
constexpr long long OFF_VR  = OFF_V   + cXE * 2;      // vflat f16 [B,H,A,E*VU]
constexpr long long OFF_Q   = OFF_VR  + cXE * 2;      // q f16 [B,H,A,32] (max A=256)
constexpr long long OFF_K   = OFF_Q   + 2097152;
constexpr long long OFF_E   = OFF_K   + 2097152;      // energy fp32 [B,H,A,A] (max 8.39M)
constexpr long long OFF_AT  = OFF_E   + 33554432;     // attn f16
constexpr long long OFF_OSP = OFF_AT  + 16777216;     // spatial attn out f16 [B,H,N,T*8]
constexpr long long OFF_OTP = OFF_OSP + cXE * 2;      // temporal attn out f16 [B,H,T,N*8]
constexpr long long OFF_X0  = OFF_OTP + cXE * 2;      // concat f16 [B,T,N,2C]
constexpr long long OFF_HH  = OFF_X0  + cXE * 4;      // FFN hidden f16 [BTN,256]
constexpr long long OFF_WH  = OFF_HH  + cBTN * 256 * 2; // f16 weights (per layer)
constexpr long long OFF_RED = OFF_WH  + 12632064LL * 2; // reduction scratch

// f16 weight sub-offsets (in halves) inside OFF_WH
constexpr long long WH_WVS = 0;         // C*C        = 4096
constexpr long long WH_WKS = 4096;      // 16384*256  = 4194304
constexpr long long WH_WQS = 4198400;
constexpr long long WH_WVT = 8392704;   // 4096
constexpr long long WH_WKT = 8396800;   // 8192*256   = 2097152
constexpr long long WH_WQT = 10493952;
constexpr long long WH_WR  = 12591104;  // 128*64     = 8192
constexpr long long WH_W1  = 12599296;  // 64*256     = 16384
constexpr long long WH_W2  = 12615680;  // 256*64     = 16384

// ================= generic WMMA GEMM =================
// C[m,n] = scale * sum_k A[m,k]*B[k,n]  (+diag mask, +bias, +relu, +residual)
// All divisors are powers of two -> passed as SHIFT COUNTS (no integer division).
struct GemmParams {
  const _Float16* A;
  const _Float16* B;
  float*    Cf;           // optional fp32 output
  _Float16* Ch;           // optional f16 output
  const float* bias;      // optional, indexed by global col
  const float* resid;     // optional, same addressing as C
  long long M, N, K;
  // A row base = (m >> a_idiv_sh)*a_ostr + (m & mask)*a_istr
  // A elem     = base + (k >> a_chunk_sh)*a_cstr + (k & chunk_mask)
  long long a_istr, a_ostr, a_cstr;
  int a_idiv_sh, a_chunk_sh;
  // B matrix base = (m0 >> b_brows_sh)*b_mstr; elem = base + k*b_ld + n (or n*b_ld + k if trans)
  long long b_ld, b_mstr;
  int b_brows_sh;
  // C offset = (m >> c_idiv_sh)*c_ostr + (m & mask)*c_istr + (n >> c_cchunk_sh)*c_ccstr + (n & cmask)
  long long c_istr, c_ostr, c_ccstr;
  int c_idiv_sh, c_cchunk_sh;
  long long diag_msk;     // if !=0 and (m & diag_msk)==n -> value = -1e9
  float scale;
  int b_trans;
  int relu;
};

#define BM 64
#define BN 64
#define BKK 32
#define LDA_S 48
#define LDB_S 48   // Bs is stored TRANSPOSED: Bs[n][k], 64 x 32 (+pad)

__global__ __launch_bounds__(128) void k_gemm(GemmParams p) {
  __shared__ __align__(16) _Float16 As[BM * LDA_S];
  __shared__ __align__(16) _Float16 Bs[BN * LDB_S];

  const int tid  = threadIdx.x;
  const int lane = tid & 31;
  const int wave = tid >> 5;
  const long long m0 = (long long)blockIdx.y * BM;
  const long long n0 = (long long)blockIdx.x * BN;
  const int wm = (wave >> 1) * 32;
  const int wn = (wave & 1) * 32;

  const long long a_imask = (1LL << p.a_idiv_sh) - 1;
  const long long a_cmask = (1LL << p.a_chunk_sh) - 1;

  v8f acc[2][2] = {};
  const long long bbase = (m0 >> p.b_brows_sh) * p.b_mstr;

  for (long long k0 = 0; k0 < p.K; k0 += BKK) {
    // ---- stage A tile (64 x 32) : 128 threads x 16 halves, vectorized ----
    {
      const int r  = tid >> 1;
      const int ks = (tid & 1) * 16;
      const long long m = m0 + r;
      const long long rowbase = (m >> p.a_idiv_sh) * p.a_ostr + (m & a_imask) * p.a_istr;
      const long long k = k0 + ks;
      const long long off = rowbase + (k >> p.a_chunk_sh) * p.a_cstr + (k & a_cmask);
      const uint4* src = (const uint4*)(p.A + off);
      uint4* dst = (uint4*)(&As[r * LDA_S + ks]);
      dst[0] = src[0];
      dst[1] = src[1];
      const long long kn = k + BKK;
      if (kn < p.K) {
        const long long off2 = rowbase + (kn >> p.a_chunk_sh) * p.a_cstr + (kn & a_cmask);
        __builtin_prefetch(p.A + off2, 0, 0);
      }
    }
    // ---- stage B tile, TRANSPOSED into LDS: Bs[n][k] ----
    if (!p.b_trans) {
      // global rows contiguous along n: vector load, scalar transpose-store
      const int r  = tid >> 2;             // k row 0..31
      const int ns = (tid & 3) * 16;       // n seg
      const long long off = bbase + (k0 + r) * p.b_ld + n0 + ns;
      const v8h va = ((const v8h*)(p.B + off))[0];
      const v8h vb = ((const v8h*)(p.B + off))[1];
#pragma unroll
      for (int i = 0; i < 8; ++i) Bs[(ns + i) * LDB_S + r]     = va[i];
#pragma unroll
      for (int i = 0; i < 8; ++i) Bs[(ns + 8 + i) * LDB_S + r] = vb[i];
    } else {
      // B^T: global contiguous along k -> fully vector path
      const int n  = tid >> 1;             // 0..63
      const int ks = (tid & 1) * 16;
      const long long off = bbase + (n0 + n) * p.b_ld + k0 + ks;
      const uint4* src = (const uint4*)(p.B + off);
      uint4* dst = (uint4*)(&Bs[n * LDB_S + ks]);
      dst[0] = src[0];
      dst[1] = src[1];
    }
    __syncthreads();

    // ---- fragments per ISA 7.12.2 (wave32), two ds_load_b128 each ----
    const int mrow = lane & 15, hi = lane >> 4;
    v16h af[2], bf[2];
#pragma unroll
    for (int fm = 0; fm < 2; ++fm) {
      const _Float16* base = &As[(wm + fm * 16 + mrow) * LDA_S + hi * 8];
      const v8h lo = ((const v8h*)base)[0];          // k = hi*8 + 0..7
      const v8h hv = *((const v8h*)(base + 16));     // k = 16 + hi*8 + 0..7
      af[fm] = __builtin_shufflevector(lo, hv, 0, 1, 2, 3, 4, 5, 6, 7,
                                       8, 9, 10, 11, 12, 13, 14, 15);
    }
#pragma unroll
    for (int fn = 0; fn < 2; ++fn) {
      const _Float16* base = &Bs[(wn + fn * 16 + (lane & 15)) * LDB_S + hi * 8];
      const v8h lo = ((const v8h*)base)[0];          // k = hi*8 + 0..7
      const v8h hv = *((const v8h*)(base + 16));     // k = 16 + hi*8 + 0..7
      bf[fn] = __builtin_shufflevector(lo, hv, 0, 1, 2, 3, 4, 5, 6, 7,
                                       8, 9, 10, 11, 12, 13, 14, 15);
    }
#pragma unroll
    for (int fm = 0; fm < 2; ++fm)
#pragma unroll
      for (int fn = 0; fn < 2; ++fn)
        acc[fm][fn] = __builtin_amdgcn_wmma_f32_16x16x32_f16(
            false, af[fm], false, bf[fn], (short)0, acc[fm][fn], false, false);
    __syncthreads();
  }

  // ---- epilogue (shift/mask addressing only) ----
  const long long c_imask = (1LL << p.c_idiv_sh) - 1;
  const long long c_cmask = (1LL << p.c_cchunk_sh) - 1;
  const int ncol = lane & 15, hi = lane >> 4;
#pragma unroll
  for (int fm = 0; fm < 2; ++fm)
#pragma unroll
    for (int fn = 0; fn < 2; ++fn)
#pragma unroll
      for (int r = 0; r < 8; ++r) {
        const long long mg = m0 + wm + fm * 16 + hi * 8 + r;
        const long long ng = n0 + wn + fn * 16 + ncol;
        float v = acc[fm][fn][r] * p.scale;
        if (p.diag_msk && (mg & p.diag_msk) == ng) v = -1e9f;
        if (p.bias) v += p.bias[ng];
        if (p.relu) v = v > 0.f ? v : 0.f;
        const long long co = (mg >> p.c_idiv_sh) * p.c_ostr + (mg & c_imask) * p.c_istr +
                             (ng >> p.c_cchunk_sh) * p.c_ccstr + (ng & c_cmask);
        if (p.resid) v += p.resid[co];
        if (p.Cf) p.Cf[co] = v;
        if (p.Ch) p.Ch[co] = (_Float16)v;
      }
}

// ================= elementwise / reduction kernels =================

__global__ void k_lift(const float* __restrict__ vals, const float* __restrict__ w,
                       const float* __restrict__ bia, float* __restrict__ x) {
  long long i = (long long)blockIdx.x * blockDim.x + threadIdx.x;
  if (i >= cXE) return;
  const int c = (int)(i & 63);
  x[i] = vals[i >> 6] * w[c] + bia[c];
}

__global__ void k_cast(const float* __restrict__ s, _Float16* __restrict__ d, long long n) {
  long long i = (long long)blockIdx.x * blockDim.x + threadIdx.x;
  if (i < n) d[i] = (_Float16)s[i];
}

// layernorm over C=64, one wave per row (wave32: 2 elems/lane), f16 out
__global__ void k_layernorm(const float* __restrict__ x, const float* __restrict__ g,
                            const float* __restrict__ b, _Float16* __restrict__ out,
                            long long rows) {
  const long long wid = (((long long)blockIdx.x * blockDim.x) + threadIdx.x) >> 5;
  const int lane = threadIdx.x & 31;
  if (wid >= rows) return;
  const float2 v = ((const float2*)(x + wid * 64))[lane];
  float s = v.x + v.y;
#pragma unroll
  for (int i = 16; i > 0; i >>= 1) s += __shfl_xor(s, i, 32);
  const float mu = s * (1.f / 64.f);
  const float dx = v.x - mu, dy = v.y - mu;
  float q = dx * dx + dy * dy;
#pragma unroll
  for (int i = 16; i > 0; i >>= 1) q += __shfl_xor(q, i, 32);
  const float rstd = rsqrtf(q * (1.f / 64.f) + 1e-5f);
  const int c = lane * 2;
  out[wid * 64 + c]     = (_Float16)(dx * rstd * g[c] + b[c]);
  out[wid * 64 + c + 1] = (_Float16)(dy * rstd * g[c + 1] + b[c + 1]);
}

// softmax over rows of length A (128 or 256), one wave per row, f16 out
__global__ void k_softmax(const float* __restrict__ e, _Float16* __restrict__ attn,
                          int A, long long rows) {
  const long long wid = (((long long)blockIdx.x * blockDim.x) + threadIdx.x) >> 5;
  const int lane = threadIdx.x & 31;
  if (wid >= rows) return;
  const float* rp = e + wid * (long long)A;
  const int per = A >> 5;
  float v[8];
  float mx = -3e38f;
  for (int i = 0; i < per; ++i) { v[i] = rp[lane + 32 * i]; mx = fmaxf(mx, v[i]); }
#pragma unroll
  for (int i = 16; i > 0; i >>= 1) mx = fmaxf(mx, __shfl_xor(mx, i, 32));
  float s = 0.f;
  for (int i = 0; i < per; ++i) { v[i] = __expf(v[i] - mx); s += v[i]; }
#pragma unroll
  for (int i = 16; i > 0; i >>= 1) s += __shfl_xor(s, i, 32);
  const float inv = 1.f / s;
  for (int i = 0; i < per; ++i) attn[wid * (long long)A + lane + 32 * i] = (_Float16)(v[i] * inv);
}

// vflat spatial: vsp[b,h,n,t*8+vu] = v[b,t,n,h*8+vu]
__global__ void k_make_vsp(const _Float16* __restrict__ v, _Float16* __restrict__ vsp) {
  long long i = (long long)blockIdx.x * blockDim.x + threadIdx.x;
  if (i >= cXE) return;
  const int vu = (int)(i & 7);
  long long r = i >> 3;
  const int t = (int)(r % cT); r /= cT;
  const int n = (int)(r % cN); r /= cN;
  const int h = (int)(r % cH);
  const long long b = r / cH;
  vsp[i] = v[(((b * cT + t) * cN + n) * cC) + h * 8 + vu];
}

// vflat temporal: vtp[b,h,t,n*8+vu] = v[b,t,n,h*8+vu]
__global__ void k_make_vtp(const _Float16* __restrict__ v, _Float16* __restrict__ vtp) {
  long long i = (long long)blockIdx.x * blockDim.x + threadIdx.x;
  if (i >= cXE) return;
  const int vu = (int)(i & 7);
  long long r = i >> 3;
  const int n = (int)(r % cN); r /= cN;
  const int t = (int)(r % cT); r /= cT;
  const int h = (int)(r % cH);
  const long long b = r / cH;
  vtp[i] = v[(((b * cT + t) * cN + n) * cC) + h * 8 + vu];
}

// x0[b,t,n, 0:64]=sp ; [64:128]=tp
__global__ void k_build_x0(const _Float16* __restrict__ sp, const _Float16* __restrict__ tp,
                           _Float16* __restrict__ x0) {
  long long i = (long long)blockIdx.x * blockDim.x + threadIdx.x;
  if (i >= cBTN * 128) return;
  const int c = (int)(i & 127);
  long long r = i >> 7;
  const int n = (int)(r % cN);
  const int t = (int)((r / cN) % cT);
  const long long b = r / (cN * cT);
  if (c < 64) {
    const int h = c >> 3, vu = c & 7;
    x0[i] = sp[(((b * cH + h) * cN + n) * (cT * 8)) + t * 8 + vu];
  } else {
    const int c2 = c - 64, h = c2 >> 3, vu = c2 & 7;
    x0[i] = tp[(((b * cH + h) * cT + t) * (cN * 8)) + n * 8 + vu];
  }
}

// conv2 + masked SE partials (per-batch)
__global__ void k_output(const float* __restrict__ x, const float* __restrict__ w,
                         const float* __restrict__ bsc, const float* __restrict__ vals,
                         const float* __restrict__ masks, float* __restrict__ out,
                         float* __restrict__ se, float* __restrict__ ms) {
  const long long r = (long long)blockIdx.x * blockDim.x + threadIdx.x;
  if (r >= cBTN) return;
  const float4* xp = (const float4*)(x + r * 64);
  const float4* wp = (const float4*)w;
  float s = 0.f;
#pragma unroll
  for (int i = 0; i < 16; ++i) {
    const float4 a = xp[i], bb = wp[i];
    s += a.x * bb.x + a.y * bb.y + a.z * bb.z + a.w * bb.w;
  }
  s += bsc[0];
  out[r] = s;
  const float d = s - vals[r];
  float e = masks[r] * d * d;
  float m = masks[r];
#pragma unroll
  for (int i = 16; i > 0; i >>= 1) { e += __shfl_xor(e, i, 32); m += __shfl_xor(m, i, 32); }
  if ((threadIdx.x & 31) == 0) {
    const int b = (int)(r / (cT * cN));
    atomicAdd(&se[b], e);
    atomicAdd(&ms[b], m);
  }
}

__global__ void k_loss(const float* __restrict__ se, const float* __restrict__ ms,
                       const float* __restrict__ it, float* __restrict__ loss) {
  if (threadIdx.x == 0) {
    float num = 0.f, den = 0.f;
    for (int b = 0; b < 16; ++b) { num += (se[b] / ms[b]) * it[b]; den += it[b]; }
    loss[0] = num / (den + 1e-5f);
  }
}

// ================= host side =================
static inline GemmParams gp0() {
  GemmParams p{};
  p.a_idiv_sh = 40; p.a_ostr = 0; p.a_chunk_sh = 40; p.a_cstr = 0;
  p.b_brows_sh = 40; p.b_mstr = 0; p.b_trans = 0;
  p.c_idiv_sh = 40; p.c_ostr = 0; p.c_cchunk_sh = 40; p.c_ccstr = 0;
  p.scale = 1.f; p.diag_msk = 0; p.relu = 0;
  p.bias = nullptr; p.resid = nullptr; p.Cf = nullptr; p.Ch = nullptr;
  return p;
}
static inline void launch_gemm(hipStream_t s, const GemmParams& p) {
  dim3 g((unsigned)(p.N / BN), (unsigned)(p.M / BM));
  k_gemm<<<g, 128, 0, s>>>(p);
}
static inline unsigned blks(long long n, int b) { return (unsigned)((n + b - 1) / b); }

extern "C" void kernel_launch(void* const* d_in, const int* in_sizes, int n_in,
                              void* d_out, int out_size, void* d_ws, size_t ws_size,
                              hipStream_t stream) {
  char* ws = (char*)d_ws;
  auto F32 = [&](long long off) { return (float*)(ws + off); };
  auto F16 = [&](long long off) { return (_Float16*)(ws + off); };

  const float* in_values = (const float*)d_in[0];
  const float* in_masks  = (const float*)d_in[1];
  const float* in_istr   = (const float*)d_in[2];
  const float* conv1_w   = (const float*)d_in[3];
  const float* conv1_b   = (const float*)d_in[4];
  const float* conv2_w   = (const float*)d_in[5];
  const float* conv2_b   = (const float*)d_in[6];
  const float* ln1_g     = (const float*)d_in[7];
  const float* ln1_b     = (const float*)d_in[8];
  const float* ln2_g     = (const float*)d_in[9];
  const float* ln2_b     = (const float*)d_in[10];
  const float* Wv_s      = (const float*)d_in[11];
  const float* Wk_s      = (const float*)d_in[12];
  const float* Wq_s      = (const float*)d_in[13];
  const float* Wv_t      = (const float*)d_in[14];
  const float* Wk_t      = (const float*)d_in[15];
  const float* Wq_t      = (const float*)d_in[16];
  const float* Wr        = (const float*)d_in[17];
  const float* br        = (const float*)d_in[18];
  const float* W1        = (const float*)d_in[19];
  const float* b1        = (const float*)d_in[20];
  const float* W2        = (const float*)d_in[21];
  const float* b2        = (const float*)d_in[22];

  float*     x    = F32(OFF_X);
  float*     x1   = F32(OFF_X1);
  _Float16*  xh   = F16(OFF_XH);
  _Float16*  v    = F16(OFF_V);
  _Float16*  vr   = F16(OFF_VR);
  _Float16*  qh   = F16(OFF_Q);
  _Float16*  kh   = F16(OFF_K);
  float*     en   = F32(OFF_E);
  _Float16*  at   = F16(OFF_AT);
  _Float16*  osp  = F16(OFF_OSP);
  _Float16*  otp  = F16(OFF_OTP);
  _Float16*  x0h  = F16(OFF_X0);
  _Float16*  hh   = F16(OFF_HH);
  _Float16*  wh   = F16(OFF_WH);
  float*     se   = F32(OFF_RED);
  float*     msum = se + 16;

  const float inv_sqrt_hd = 0.17677669529663687f; // 1/sqrt(32)
  float* outp = (float*)d_out;

  k_lift<<<blks(cXE, 256), 256, 0, stream>>>(in_values, conv1_w, conv1_b, x);

  for (int L = 0; L < 4; ++L) {
    struct { const float* s; long long off, n; } casts[] = {
      { Wv_s + (long long)L * 4096,    WH_WVS, 4096 },
      { Wk_s + (long long)L * 4194304, WH_WKS, 4194304 },
      { Wq_s + (long long)L * 4194304, WH_WQS, 4194304 },
      { Wv_t + (long long)L * 4096,    WH_WVT, 4096 },
      { Wk_t + (long long)L * 2097152, WH_WKT, 2097152 },
      { Wq_t + (long long)L * 2097152, WH_WQT, 2097152 },
      { Wr   + (long long)L * 8192,    WH_WR,  8192 },
      { W1   + (long long)L * 16384,   WH_W1,  16384 },
      { W2   + (long long)L * 16384,   WH_W2,  16384 },
    };
    for (auto& cst : casts)
      k_cast<<<blks(cst.n, 256), 256, 0, stream>>>(cst.s, wh + cst.off, cst.n);

    // ---- ln1 ----
    k_layernorm<<<blks(cBTN * 32, 256), 256, 0, stream>>>(x, ln1_g + L * 64, ln1_b + L * 64, xh, cBTN);

    // ======== spatial attention (A=N=128, E=T=256) ========
    { // v = xn @ Wv_s
      GemmParams p = gp0();
      p.A = xh; p.a_istr = 64; p.B = wh + WH_WVS; p.b_ld = 64;
      p.Ch = v; p.c_istr = 64; p.M = cBTN; p.N = 64; p.K = 64;
      launch_gemm(stream, p);
    }
    k_make_vsp<<<blks(cXE, 256), 256, 0, stream>>>(v, vr);
    for (int qk = 0; qk < 2; ++qk) { // [2048 x 16384] x [16384 x 256]
      GemmParams p = gp0();
      p.A = xh; p.a_idiv_sh = 7; p.a_istr = 64; p.a_ostr = cT * cN * 64;
      p.a_chunk_sh = 6; p.a_cstr = cN * 64;
      p.B = wh + (qk ? WH_WKS : WH_WQS); p.b_ld = 256;
      p.Ch = qk ? kh : qh;
      p.c_idiv_sh = 7; p.c_istr = 32; p.c_ostr = cH * cN * 32;     // -> [B,H,N,32]
      p.c_cchunk_sh = 5; p.c_ccstr = cN * 32;
      p.M = cB * cN; p.N = 256; p.K = cT * 64;
      launch_gemm(stream, p);
    }
    { // energy = q k^T / sqrt(hd), diag mask ; batched over (b,h)
      GemmParams p = gp0();
      p.A = qh; p.a_istr = 32;
      p.B = kh; p.b_trans = 1; p.b_ld = 32; p.b_brows_sh = 7; p.b_mstr = cN * 32;
      p.Cf = en; p.c_istr = cN;
      p.M = cB * cH * cN; p.N = cN; p.K = 32;
      p.scale = inv_sqrt_hd; p.diag_msk = cN - 1;
      launch_gemm(stream, p);
    }
    k_softmax<<<blks(cB * cH * cN * 32, 256), 256, 0, stream>>>(en, at, (int)cN, cB * cH * cN);
    { // out = attn @ vflat : [16384 x 128] x [128 x 2048] per (b,h)
      GemmParams p = gp0();
      p.A = at; p.a_istr = cN;
      p.B = vr; p.b_ld = cT * 8; p.b_brows_sh = 7; p.b_mstr = cN * cT * 8;
      p.Ch = osp; p.c_istr = cT * 8;
      p.M = cB * cH * cN; p.N = cT * 8; p.K = cN;
      launch_gemm(stream, p);
    }

    // ======== temporal attention (A=T=256, E=N=128) ========
    { // v = xn @ Wv_t
      GemmParams p = gp0();
      p.A = xh; p.a_istr = 64; p.B = wh + WH_WVT; p.b_ld = 64;
      p.Ch = v; p.c_istr = 64; p.M = cBTN; p.N = 64; p.K = 64;
      launch_gemm(stream, p);
    }
    k_make_vtp<<<blks(cXE, 256), 256, 0, stream>>>(v, vr);
    for (int qk = 0; qk < 2; ++qk) { // [4096 x 8192] x [8192 x 256]
      GemmParams p = gp0();
      p.A = xh; p.a_istr = cN * 64;                        // rows (b,t), contiguous K
      p.B = wh + (qk ? WH_WKT : WH_WQT); p.b_ld = 256;
      p.Ch = qk ? kh : qh;
      p.c_idiv_sh = 8; p.c_istr = 32; p.c_ostr = cH * cT * 32;   // -> [B,H,T,32]
      p.c_cchunk_sh = 5; p.c_ccstr = cT * 32;
      p.M = cB * cT; p.N = 256; p.K = cN * 64;
      launch_gemm(stream, p);
    }
    {
      GemmParams p = gp0();
      p.A = qh; p.a_istr = 32;
      p.B = kh; p.b_trans = 1; p.b_ld = 32; p.b_brows_sh = 8; p.b_mstr = cT * 32;
      p.Cf = en; p.c_istr = cT;
      p.M = cB * cH * cT; p.N = cT; p.K = 32;
      p.scale = inv_sqrt_hd; p.diag_msk = cT - 1;
      launch_gemm(stream, p);
    }
    k_softmax<<<blks(cB * cH * cT * 32, 256), 256, 0, stream>>>(en, at, (int)cT, cB * cH * cT);
    { // [32768 x 256] x [256 x 1024] per (b,h)
      GemmParams p = gp0();
      p.A = at; p.a_istr = cT;
      p.B = vr; p.b_ld = cN * 8; p.b_brows_sh = 8; p.b_mstr = cT * cN * 8;
      p.Ch = otp; p.c_istr = cN * 8;
      p.M = cB * cH * cT; p.N = cN * 8; p.K = cT;
      launch_gemm(stream, p);
    }

    // ---- concat + Wr + residual ----
    k_build_x0<<<blks(cBTN * 128, 256), 256, 0, stream>>>(osp, otp, x0h);
    {
      GemmParams p = gp0();
      p.A = x0h; p.a_istr = 128;
      p.B = wh + WH_WR; p.b_ld = 64;
      p.bias = br + L * 64; p.resid = x;
      p.Cf = x1; p.c_istr = 64;
      p.M = cBTN; p.N = 64; p.K = 128;
      launch_gemm(stream, p);
    }
    // ---- ln2 + FFN ----
    k_layernorm<<<blks(cBTN * 32, 256), 256, 0, stream>>>(x1, ln2_g + L * 64, ln2_b + L * 64, xh, cBTN);
    {
      GemmParams p = gp0();
      p.A = xh; p.a_istr = 64;
      p.B = wh + WH_W1; p.b_ld = 256;
      p.bias = b1 + L * 256; p.relu = 1;
      p.Ch = hh; p.c_istr = 256;
      p.M = cBTN; p.N = 256; p.K = 64;
      launch_gemm(stream, p);
    }
    {
      GemmParams p = gp0();
      p.A = hh; p.a_istr = 256;
      p.B = wh + WH_W2; p.b_ld = 64;
      p.bias = b2 + L * 64; p.resid = x1;
      p.Cf = x; p.c_istr = 64;
      p.M = cBTN; p.N = 64; p.K = 256;
      launch_gemm(stream, p);
    }
  }

  // ---- conv2 + loss ----
  hipMemsetAsync(se, 0, 32 * sizeof(float), stream);
  k_output<<<blks(cBTN, 256), 256, 0, stream>>>(x, conv2_w, conv2_b, in_values, in_masks,
                                                outp, se, msum);
  k_loss<<<1, 32, 0, stream>>>(se, msum, in_istr, outp + cBTN);
}
